// HalfInteger2BitLinear_87703232184565
// MI455X (gfx1250) — matrix operand order repaired and checked by reading, not compiled, not run on hardware
//
#include <hip/hip_runtime.h>

#define M_DIM 8192
#define N_DIM 8192
#define T_DIM 16
#define KSPLIT 4
#define SQRT_N 90.50966799187809f

typedef __attribute__((ext_vector_type(16))) _Float16 v16h;
typedef __attribute__((ext_vector_type(8)))  _Float16 v8h;
typedef __attribute__((ext_vector_type(8)))  float    v8f;
typedef __attribute__((ext_vector_type(4)))  int      v4i;

// ---------------------------------------------------------------------------
// Kernel A: xh[t][n] = fp16( FWHT(input[t]*SU)[n] / (sqrt(N)*1024) )
// One block per token row; 8192-float FWHT done in 32KB LDS.
// ---------------------------------------------------------------------------
__global__ __launch_bounds__(256) void fwht_in_kernel(
    const float* __restrict__ inp, const float* __restrict__ SU,
    _Float16* __restrict__ xh) {
  __shared__ float buf[N_DIM];
  const int t = blockIdx.x;
  const float* row = inp + (size_t)t * N_DIM;
  for (int n = threadIdx.x; n < N_DIM; n += 256)
    buf[n] = row[n] * SU[n];
  for (int h = 1; h < N_DIM; h <<= 1) {
    __syncthreads();
    for (int p = threadIdx.x; p < N_DIM / 2; p += 256) {
      int base = ((p & ~(h - 1)) << 1) | (p & (h - 1));
      float a = buf[base], b = buf[base + h];
      buf[base] = a + b;
      buf[base + h] = a - b;
    }
  }
  __syncthreads();
  const float scale = 1.0f / (SQRT_N * 1024.0f);
  _Float16* orow = xh + (size_t)t * N_DIM;
  for (int n = threadIdx.x; n < N_DIM; n += 256)
    orow[n] = (_Float16)(buf[n] * scale);
}

// ---------------------------------------------------------------------------
// Kernel B: z[t][m] = sum_n xh[t][n] * ((f16)Qidxs[m][n] - 1.5)
// One wave per (16-token x 16-m) tile per K-slice; WMMA f32_16x16x32_f16.
// 2048 waves total; each Qidxs element is read exactly once (HBM-bound).
// ---------------------------------------------------------------------------
__global__ __launch_bounds__(256) void gemm_wmma_kernel(
    const _Float16* __restrict__ xh, const int* __restrict__ Qidxs,
    float* __restrict__ zp) {
  const int wave   = (blockIdx.x * 256 + threadIdx.x) >> 5;  // 0..2047
  const int lane   = threadIdx.x & 31;
  const int mtile  = wave >> 2;    // 0..511
  const int kslice = wave & 3;     // 0..3
  const int m0     = mtile * 16;
  const int kbase  = kslice * (N_DIM / KSPLIT);
  const int nsub   = lane & 15;    // token row for A, m column for B
  const int khalf  = lane >> 4;    // K-half selector per ISA lane layout

  const _Float16* arow = xh + (size_t)nsub * N_DIM;
  const int* qrow = Qidxs + (size_t)(m0 + nsub) * N_DIM;

  v8f c = {};
  for (int k0 = kbase; k0 < kbase + N_DIM / KSPLIT; k0 += 32) {
    // A fragment (16x32 f16): lane&15 = M row; VGPR0-3 hold K +0..7(+khalf*8),
    // VGPR4-7 hold K +16..23(+khalf*8)
    v8h alo = *(const v8h*)(arow + k0 + khalf * 8);
    v8h ahi = *(const v8h*)(arow + k0 + 16 + khalf * 8);
    v16h a;
#pragma unroll
    for (int i = 0; i < 8; ++i) { a[i] = alo[i]; a[8 + i] = ahi[i]; }

    // B fragment (32x16 f16): lane&15 = N column, elems 0-7 K=+0..7(+khalf*8),
    // elems 8-15 K=+16..23(+khalf*8) -> 4x b128 loads.
    const int* qp = qrow + k0 + khalf * 8;
    v16h b;
#pragma unroll
    for (int j = 0; j < 4; ++j) {
      v4i q = *(const v4i*)(qp + (j & 1) * 4 + (j >> 1) * 16);
#pragma unroll
      for (int e = 0; e < 4; ++e)
        b[j * 4 + e] = (_Float16)(short)q[e] - (_Float16)1.5f;
    }

    c = __builtin_amdgcn_wmma_f32_16x16x32_f16(false, a, false, b,
                                               (short)0, c, false, false);
  }

  // C layout: VGPR r -> M = r + 8*(lane>>4) (= token), N = lane&15 (= m)
  float* zrow = zp + (size_t)kslice * (T_DIM * M_DIM);
  const int tbase = 8 * khalf;
#pragma unroll
  for (int r = 0; r < 8; ++r)
    zrow[(size_t)(tbase + r) * M_DIM + m0 + nsub] = c[r];
}

// ---------------------------------------------------------------------------
// Kernel C: out[t][m] = SV[m] * FWHT( sum_s zp[s][t] * Wscale*1024 )[m]/sqrt(M)
// ---------------------------------------------------------------------------
__global__ __launch_bounds__(256) void fwht_out_kernel(
    const float* __restrict__ zp, const float* __restrict__ SV,
    const float* __restrict__ Wscale, float* __restrict__ out) {
  __shared__ float buf[M_DIM];
  const int t = blockIdx.x;
  for (int m = threadIdx.x; m < M_DIM; m += 256) {
    float s = 0.f;
#pragma unroll
    for (int p = 0; p < KSPLIT; ++p)
      s += zp[(size_t)p * (T_DIM * M_DIM) + (size_t)t * M_DIM + m];
    buf[m] = s;
  }
  for (int h = 1; h < M_DIM; h <<= 1) {
    __syncthreads();
    for (int p = threadIdx.x; p < M_DIM / 2; p += 256) {
      int base = ((p & ~(h - 1)) << 1) | (p & (h - 1));
      float a = buf[base], b = buf[base + h];
      buf[base] = a + b;
      buf[base + h] = a - b;
    }
  }
  __syncthreads();
  const float fac = Wscale[0] * 1024.0f / SQRT_N;
  for (int m = threadIdx.x; m < M_DIM; m += 256)
    out[(size_t)t * M_DIM + m] = buf[m] * fac * SV[m];
}

// ---------------------------------------------------------------------------
extern "C" void kernel_launch(void* const* d_in, const int* in_sizes, int n_in,
                              void* d_out, int out_size, void* d_ws, size_t ws_size,
                              hipStream_t stream) {
  const float* inp    = (const float*)d_in[0];
  const int*   Qidxs  = (const int*)d_in[1];
  const float* SU     = (const float*)d_in[2];
  const float* SV     = (const float*)d_in[3];
  const float* Wscale = (const float*)d_in[4];
  // d_in[5] = grid (fp16) is fixed [-1.5,-0.5,0.5,1.5] -> hardcoded as q-1.5
  float* outp = (float*)d_out;

  _Float16* xh = (_Float16*)d_ws;                      // 16*8192 fp16 = 256KB
  float* zp = (float*)((char*)d_ws +
                       (size_t)T_DIM * N_DIM * sizeof(_Float16));  // 4*16*8192 f32 = 2MB

  fwht_in_kernel<<<T_DIM, 256, 0, stream>>>(inp, SU, xh);
  gemm_wmma_kernel<<<(512 * KSPLIT) / 8, 256, 0, stream>>>(xh, Qidxs, zp);
  fwht_out_kernel<<<T_DIM, 256, 0, stream>>>(zp, SV, Wscale, outp);
}
